// ExpertsChooseMaskedExpand_69157563400660
// MI455X (gfx1250) — matrix operand order repaired and checked by reading, not compiled, not run on hardware
//
#include <hip/hip_runtime.h>
#include <hip/hip_bf16.h>

// MoE experts-choose, reassociated: xd = dispatch^T@x ; z = combine@xd ;
// out = z@W^T + bias*rowsum(combine).  All GEMMs on v_wmma_f32_16x16x32_bf16.

typedef __attribute__((ext_vector_type(16))) __bf16 v16bf;
typedef __attribute__((ext_vector_type(8)))  __bf16 v8bf;
typedef __attribute__((ext_vector_type(8)))  float  v8f;
typedef __attribute__((ext_vector_type(4)))  float  f32x4;

#define B_   4
#define T_   4096
#define E_   8
#define C_   512
#define I_   128
#define O_   4096
#define K3_  1024   // E_*I_

__device__ __forceinline__ v16bf cat16(v8bf lo, v8bf hi) {
  return __builtin_shufflevector(lo, hi, 0,1,2,3,4,5,6,7,8,9,10,11,12,13,14,15);
}

__device__ __forceinline__ v16bf load16bf(const __bf16* p) {
  v8bf lo = *(const v8bf*)p;
  v8bf hi = *(const v8bf*)(p + 8);
  return cat16(lo, hi);
}

// A fragment (16x32, M x K), per-lane m fixed, fp32 source with K-stride `stride`
__device__ __forceinline__ v16bf load_a_strided(const float* base, int k, int h, int stride) {
  v16bf a;
#pragma unroll
  for (int j = 0; j < 8; ++j)
    a[j] = (__bf16)base[(size_t)(k + 16*h + j) * stride];
#pragma unroll
  for (int j = 0; j < 8; ++j)
    a[8+j] = (__bf16)base[(size_t)(k + 16*h + 8 + j) * stride];
  return a;
}

// B fragment (32x16, K x N), per-lane n fixed, fp32 source with K-stride `stride`
__device__ __forceinline__ v16bf load_b_strided(const float* base, int k, int h, int stride) {
  v16bf b;
#pragma unroll
  for (int j = 0; j < 16; ++j)
    b[j] = (__bf16)base[(size_t)(k + 16*h + j) * stride];
  return b;
}

// A fragment from K-contiguous fp32 row (p = row + kbase); also accumulates the
// raw fp32 sum of the 16 elements touched (for fused combine row-sums).
__device__ __forceinline__ v16bf load_a_contig_f32_sum(const float* p, int h, float& sum) {
  const f32x4* q0 = (const f32x4*)(p + 16*h);
  const f32x4* q1 = (const f32x4*)(p + 16*h + 8);
  f32x4 x0 = q0[0], x1 = q0[1], x2 = q1[0], x3 = q1[1];
  v16bf a;
#pragma unroll
  for (int j = 0; j < 4; ++j) {
    a[j]    = (__bf16)x0[j];
    a[4+j]  = (__bf16)x1[j];
    a[8+j]  = (__bf16)x2[j];
    a[12+j] = (__bf16)x3[j];
    sum += x0[j] + x1[j] + x2[j] + x3[j];
  }
  return a;
}

__device__ __forceinline__ v8f wmma_bf16(v16bf a, v16bf b, v8f c) {
  return __builtin_amdgcn_wmma_f32_16x16x32_bf16(false, a, false, b, (short)0, c, false, false);
}

// ---------- prep: weight fp32 -> bf16 (same flat layout: w[e][o][i]) ----------
__global__ void cvt_weight(const float* __restrict__ w, __bf16* __restrict__ wbf) {
  size_t i = (size_t)blockIdx.x * 256 + threadIdx.x;
  wbf[i] = (__bf16)w[i];
}

// ---------- K1: xd_t[b,e][i][c] = sum_t dispatch[b,t,e,c] * x[b,t,e,i] ----------
// wave tile 16(c) x 128(i): dispatch (HBM-heavy) is read exactly once.
__global__ void dispatch_gemm(const float* __restrict__ x,
                              const float* __restrict__ disp,
                              __bf16* __restrict__ xd_t) {
  const int lane = threadIdx.x & 31;
  const int wave = threadIdx.x >> 5;
  const int tile = blockIdx.x * 8 + wave;       // 1024 tiles
  const int be   = tile >> 5;                   // 32 c-tiles per (b,e)
  const int c0   = (tile & 31) * 16;
  const int b = be >> 3, e = be & 7;
  const int h = lane >> 4, l16 = lane & 15;

  const float* abase = disp + ((size_t)b * T_ * E_ + e) * C_ + (c0 + l16); // t-stride E_*C_
  const float* bbase[8];
#pragma unroll
  for (int s = 0; s < 8; ++s)
    bbase[s] = x + ((size_t)b * T_ * E_ + e) * I_ + (s*16 + l16);          // t-stride E_*I_

  v8f acc[8] = {};
  for (int k = 0; k < T_; k += 32) {
    v16bf a = load_a_strided(abase, k, h, E_*C_);
#pragma unroll
    for (int s = 0; s < 8; ++s) {
      v16bf bb = load_b_strided(bbase[s], k, h, E_*I_);
      acc[s] = wmma_bf16(a, bb, acc[s]);
    }
  }
#pragma unroll
  for (int s = 0; s < 8; ++s) {
    int n = s*16 + l16;
    __bf16* dst = xd_t + ((size_t)be * I_ + n) * C_ + c0 + 8*h;  // 8 contiguous c
    v8bf st;
#pragma unroll
    for (int r = 0; r < 8; ++r) st[r] = (__bf16)acc[s][r];
    *(v8bf*)dst = st;
  }
}

// ---------- K2: z[b,t][e*128+i] = sum_c combine[b,t,e,c] * xd[b,e,c,i] ----------
// wave tile 16(t) x 128(i): combine is read exactly once; fused per-(b,t,e)
// row sums written to Spart (each (b,t,e) owned by exactly one wave).
__global__ void combine_gemm(const float* __restrict__ comb,
                             const __bf16* __restrict__ xd_t,
                             __bf16* __restrict__ z,
                             float* __restrict__ Spart) {
  const int lane = threadIdx.x & 31;
  const int wave = threadIdx.x >> 5;
  const int tile = blockIdx.x * 8 + wave;       // 8192 tiles
  const int be   = tile >> 8;                   // 256 t-tiles per (b,e)
  const int t0   = (tile & 255) * 16;
  const int b = be >> 3, e = be & 7;
  const int h = lane >> 4, l16 = lane & 15;

  const float* arow = comb + ((size_t)(b * T_ + t0 + l16) * E_ + e) * C_;
  const __bf16* bbase[8];
#pragma unroll
  for (int s = 0; s < 8; ++s)
    bbase[s] = xd_t + ((size_t)be * I_ + s*16 + l16) * C_;

  v8f acc[8] = {};
  float asum = 0.f;
  for (int k = 0; k < C_; k += 32) {
    v16bf a = load_a_contig_f32_sum(arow + k, h, asum);
#pragma unroll
    for (int s = 0; s < 8; ++s) {
      v16bf bb = load16bf(bbase[s] + k + 16*h);
      acc[s] = wmma_bf16(a, bb, acc[s]);
    }
  }
#pragma unroll
  for (int s = 0; s < 8; ++s) {
    int n = s*16 + l16;
#pragma unroll
    for (int r = 0; r < 8; ++r) {
      int t = t0 + r + 8*h;
      z[((size_t)b * T_ + t) * K3_ + e * I_ + n] = (__bf16)acc[s][r];
    }
  }
  // row sum for row (t0+l16): lane l16 (h=0) + lane l16+16 (h=1) cover the full row
  float other = __shfl_xor(asum, 16);
  if (h == 0)
    Spart[((size_t)b * T_ + t0 + l16) * E_ + e] = asum + other;
}

// ---------- fold: S[bt] = sum_e Spart[bt][e] ----------
__global__ void fold_S(const float* __restrict__ Spart, float* __restrict__ S) {
  int bt = blockIdx.x * 256 + threadIdx.x;      // 16384 rows
  float s = 0.f;
#pragma unroll
  for (int e = 0; e < E_; ++e) s += Spart[(size_t)bt * E_ + e];
  S[bt] = s;
}

// ---------- K3: out[bt][o] = sum_k z[bt][k]*W[k][o] + bias[o]*S[bt] ----------
// wave tile 32(m) x 64(o); z (32 MB) and weights (8 MB) are L2-resident.
__global__ void out_gemm(const __bf16* __restrict__ z,
                         const __bf16* __restrict__ wbf,
                         const float* __restrict__ bias,
                         const float* __restrict__ S,
                         float* __restrict__ out) {
  const int lane = threadIdx.x & 31;
  const int wave = threadIdx.x >> 5;
  const int tile = blockIdx.x * 8 + wave;       // 32768 tiles
  const int m0   = (tile >> 6) * 32;
  const int o0   = (tile & 63) * 64;
  const int h = lane >> 4, l16 = lane & 15;

  const __bf16* arow0 = z + (size_t)(m0 + l16) * K3_;
  const __bf16* arow1 = z + (size_t)(m0 + 16 + l16) * K3_;
  v8f acc[8] = {};
  for (int k = 0; k < K3_; k += 32) {
    v16bf a0 = cat16(*(const v8bf*)(arow0 + k + 16*h),
                     *(const v8bf*)(arow0 + k + 16*h + 8));
    v16bf a1 = cat16(*(const v8bf*)(arow1 + k + 16*h),
                     *(const v8bf*)(arow1 + k + 16*h + 8));
    const int kb = k + 16*h;        // 16-aligned chunk, never crosses a 128 block
    const int e  = kb >> 7;
    const int i0 = kb & 127;
#pragma unroll
    for (int s = 0; s < 4; ++s) {
      int o = o0 + s*16 + l16;
      v16bf bb = load16bf(wbf + ((size_t)e * O_ + o) * I_ + i0);
      acc[s]     = wmma_bf16(a0, bb, acc[s]);
      acc[4 + s] = wmma_bf16(a1, bb, acc[4 + s]);
    }
  }
  float sv[16];
#pragma unroll
  for (int r = 0; r < 8; ++r) {
    sv[r]     = S[m0 + r + 8*h];
    sv[8 + r] = S[m0 + 16 + r + 8*h];
  }
#pragma unroll
  for (int s = 0; s < 4; ++s) {
    int o = o0 + s*16 + l16;
    float bo = bias[o];
#pragma unroll
    for (int r = 0; r < 8; ++r) {
      out[(size_t)(m0 + r + 8*h) * O_ + o]      = acc[s][r]     + bo * sv[r];
      out[(size_t)(m0 + 16 + r + 8*h) * O_ + o] = acc[4 + s][r] + bo * sv[8 + r];
    }
  }
}

extern "C" void kernel_launch(void* const* d_in, const int* in_sizes, int n_in,
                              void* d_out, int out_size, void* d_ws, size_t ws_size,
                              hipStream_t stream) {
  const float* x      = (const float*)d_in[0];
  const float* weight = (const float*)d_in[1];
  const float* bias   = (const float*)d_in[2];
  const float* comb   = (const float*)d_in[3];
  const float* disp   = (const float*)d_in[4];
  float* out = (float*)d_out;

  char* ws = (char*)d_ws;
  __bf16* wbf   = (__bf16*)(ws);                                  //  8 MiB
  __bf16* xd_t  = (__bf16*)(ws + (8u << 20));                     //  4 MiB
  __bf16* z     = (__bf16*)(ws + (12u << 20));                    // 32 MiB
  float*  S     = (float*)(ws + (44u << 20));                     // 64 KiB
  float*  Spart = (float*)(ws + (44u << 20) + (64u << 10));       // 512 KiB

  cvt_weight   <<<16384, 256, 0, stream>>>(weight, wbf);          // 4096*1024 elems
  dispatch_gemm<<<128,   256, 0, stream>>>(x, disp, xd_t);        // 1024 wave-tiles
  combine_gemm <<<1024,  256, 0, stream>>>(comb, xd_t, z, Spart); // 8192 wave-tiles
  fold_S       <<<64,    256, 0, stream>>>(Spart, S);
  out_gemm     <<<4096,  256, 0, stream>>>(z, wbf, bias, S, out); // 32768 wave-tiles
}
